// GCNModel_10986526343322
// MI455X (gfx1250) — matrix operand order repaired and checked by reading, compiled, not run on hardware
//
#include <hip/hip_runtime.h>

// ---------------------------------------------------------------------------
// CDNA5 / gfx1250 GCN forward:  3x [WMMA-GEMM -> self-loop init -> edge
// scatter-add (f32 atomics) -> bias+ReLU] then WMMA MLP head.
// Wave32; WMMA bf16 16x16x32, f32 accumulation. B pre-packed to fragment
// layout so the GEMM inner loop is pure b128 loads + v_wmma.
// ---------------------------------------------------------------------------

typedef __attribute__((ext_vector_type(16))) __bf16 v16bf;
typedef __attribute__((ext_vector_type(8)))  __bf16 v8bf;
typedef __attribute__((ext_vector_type(8)))  float  v8f;

#define DHID 64   // GCN inner dim

// ------------------------------- small utils -------------------------------

__global__ void k_f2bf(const float* __restrict__ src, __bf16* __restrict__ dst,
                       long n) {
  long i = (long)blockIdx.x * blockDim.x + threadIdx.x;
  if (i < n) dst[i] = (__bf16)src[i];
}

// Repack W[K x Nreal] (f32, row-major) into WMMA B-fragment-major bf16:
//   out[((ntile*(K/32) + kk32)*32 + lane)*16 + i] = W[srcK, srcN]   (0 if OOB)
//   srcK = kk32*32 + ((i>>3)<<4) + ((lane>>4)<<3) + (i&7) ; srcN = ntile*16 + (lane&15)
// Total threads = K * Npad.
__global__ void k_packB(const float* __restrict__ W, __bf16* __restrict__ out,
                        int K, int Npad, int Nreal) {
  int t = blockIdx.x * blockDim.x + threadIdx.x;
  if (t >= K * Npad) return;
  int i     = t & 15;
  int lane  = (t >> 4) & 31;
  int rest  = t >> 9;
  int k32   = K >> 5;
  int kk32  = rest % k32;
  int ntile = rest / k32;
  int srcK  = (kk32 << 5) + ((i >> 3) << 4) + ((lane >> 4) << 3) + (i & 7);
  int srcN  = (ntile << 4) + (lane & 15);
  float v = (srcN < Nreal) ? W[(size_t)srcK * Nreal + srcN] : 0.0f;
  out[t] = (__bf16)v;
}

__global__ void k_deg_init(float* __restrict__ deg, int n) {
  int i = blockIdx.x * blockDim.x + threadIdx.x;
  if (i < n) deg[i] = 1.0f;   // self-loop weight
}

__global__ void k_deg_accum(const int* __restrict__ dst_idx,
                            const float* __restrict__ ew,
                            float* __restrict__ deg, int E) {
  int e = blockIdx.x * blockDim.x + threadIdx.x;
  if (e < E) atomicAdd(&deg[dst_idx[e]], ew[e]);
}

__global__ void k_dinv(const float* __restrict__ deg, float* __restrict__ dinv,
                       float* __restrict__ selfn, int n) {
  int i = blockIdx.x * blockDim.x + threadIdx.x;
  if (i >= n) return;
  float r = rsqrtf(fmaxf(deg[i], 1e-12f));
  dinv[i] = r;
  selfn[i] = r * r;
}

__global__ void k_norm(const int* __restrict__ src_idx,
                       const int* __restrict__ dst_idx,
                       const float* __restrict__ ew,
                       const float* __restrict__ dinv,
                       float* __restrict__ nrm, int E) {
  int e = blockIdx.x * blockDim.x + threadIdx.x;
  if (e < E) nrm[e] = dinv[src_idx[e]] * ew[e] * dinv[dst_idx[e]];
}

// agg[i,:] = Hw[i,:] * selfnorm[i]
__global__ void k_selfinit(const float* __restrict__ Hw,
                           const float* __restrict__ selfn,
                           float* __restrict__ agg, long nd) {
  long t = (long)blockIdx.x * blockDim.x + threadIdx.x;
  if (t >= nd) return;
  long i = t >> 6;               // D = 64
  agg[t] = Hw[t] * selfn[i];
}

// One thread per (edge, 4-feature chunk): float4 gather, 4 f32 atomics.
__global__ void k_scatter(const float* __restrict__ Hw,
                          const float* __restrict__ nrm,
                          const int* __restrict__ src_idx,
                          const int* __restrict__ dst_idx,
                          float* __restrict__ agg, int E) {
  long t = (long)blockIdx.x * blockDim.x + threadIdx.x;
  if (t >= (long)E * 16) return;          // D/4 = 16 chunks per edge
  int e = (int)(t >> 4);
  int c = (int)((t & 15) << 2);
  float w = nrm[e];
  const float4 v = *(const float4*)(Hw + (size_t)src_idx[e] * DHID + c);
  float* a = agg + (size_t)dst_idx[e] * DHID + c;
  atomicAdd(a + 0, v.x * w);
  atomicAdd(a + 1, v.y * w);
  atomicAdd(a + 2, v.z * w);
  atomicAdd(a + 3, v.w * w);
}

// h_bf[i,f] = bf16(relu(agg[i,f] + b[f]))
__global__ void k_biasrelu(const float* __restrict__ agg,
                           const float* __restrict__ b,
                           __bf16* __restrict__ hb, long nd) {
  long t = (long)blockIdx.x * blockDim.x + threadIdx.x;
  if (t >= nd) return;
  int f = (int)(t & (DHID - 1));
  hb[t] = (__bf16)fmaxf(agg[t] + b[f], 0.0f);
}

// ------------------------------- WMMA GEMM ---------------------------------
// C[M x Nreal] = A[M x K](bf16,row-major) @ B(fragment-packed bf16, Npad cols)
// One wave computes a 16 x (16*NT) strip: A fragment loaded once per k-step,
// NT WMMAs issued against NT accumulators. 4 waves (4 M-tiles) per block.
template <int NT>
__global__ void k_gemm(const __bf16* __restrict__ A, const __bf16* __restrict__ Bf,
                       const float* __restrict__ bias,
                       float* __restrict__ C, __bf16* __restrict__ Cbf,
                       int M, int K, int Npad, int Nreal, int ldc, int relu) {
  const int lane  = threadIdx.x & 31;
  const int wave  = threadIdx.x >> 5;
  const int mtile = blockIdx.x * 4 + wave;
  const int nt0   = blockIdx.y * NT;      // first n-tile of this wave's strip
  if (mtile * 16 >= M) return;            // wave-uniform exit (EXEC stays full)

  const int row  = mtile * 16 + (lane & 15);        // A row for this lane
  const int acol = (lane >> 4) << 3;                // 0 or 8 (K sub-offset)
  const int k32  = K >> 5;

  v8f acc[NT];
  #pragma unroll
  for (int nt = 0; nt < NT; ++nt) acc[nt] = (v8f){};

  const __bf16* Arow = A + (size_t)row * K;
  for (int kk32 = 0; kk32 < k32; ++kk32) {
    const int kk = kk32 << 5;
    // A fragment: 16-bit A 16x32 layout (ISA 7.12.2); two contiguous 16B loads
    v8bf a0 = *(const v8bf*)(Arow + kk + acol);
    v8bf a1 = *(const v8bf*)(Arow + kk + 16 + acol);
    v16bf a = __builtin_shufflevector(a0, a1, 0, 1, 2, 3, 4, 5, 6, 7,
                                      8, 9, 10, 11, 12, 13, 14, 15);
    #pragma unroll
    for (int nt = 0; nt < NT; ++nt) {
      // B fragment: contiguous 32B per lane in packed layout
      const v16bf b = *(const v16bf*)(
          Bf + ((((size_t)(nt0 + nt) * k32 + kk32) << 5) + lane) * 16);
      acc[nt] = __builtin_amdgcn_wmma_f32_16x16x32_bf16(
          false, a, false, b, (short)0, acc[nt], false, false);
    }
  }

  const int mbase = mtile * 16 + ((lane >> 4) << 3);
  #pragma unroll
  for (int nt = 0; nt < NT; ++nt) {
    const int n = (nt0 + nt) * 16 + (lane & 15);
    if (n < Nreal) {
      const float bi = bias ? bias[n] : 0.0f;
      #pragma unroll
      for (int r = 0; r < 8; ++r) {
        float v = acc[nt][r] + bi;
        if (relu) v = fmaxf(v, 0.0f);
        const size_t o = (size_t)(mbase + r) * ldc + n;
        if (C)   C[o]   = v;
        if (Cbf) Cbf[o] = (__bf16)v;
      }
    }
  }
}

// ------------------------------- launcher ----------------------------------

extern "C" void kernel_launch(void* const* d_in, const int* in_sizes, int n_in,
                              void* d_out, int out_size, void* d_ws, size_t ws_size,
                              hipStream_t stream) {
  const float* x   = (const float*)d_in[0];
  const int*   ei  = (const int*)  d_in[1];
  const float* ew  = (const float*)d_in[2];
  const float* W0  = (const float*)d_in[3];
  const float* b0  = (const float*)d_in[4];
  const float* W1  = (const float*)d_in[5];
  const float* b1  = (const float*)d_in[6];
  const float* W2  = (const float*)d_in[7];
  const float* b2  = (const float*)d_in[8];
  const float* Wd1 = (const float*)d_in[9];
  const float* bd1 = (const float*)d_in[10];
  const float* Wd2 = (const float*)d_in[11];
  const float* bd2 = (const float*)d_in[12];

  const int FIN = 128, HMLP = 256;
  const int N = in_sizes[0] / FIN;
  const int E = in_sizes[2];
  const int* srcI = ei;
  const int* dstI = ei + E;

  // ---- workspace carve (256B aligned regions) ----
  char* ws = (char*)d_ws;
  size_t off = 0;
  auto carve = [&](size_t bytes) -> void* {
    void* p = ws + off;
    off += (bytes + 255) & ~(size_t)255;
    return p;
  };
  float*  Hw    = (float*) carve((size_t)N * DHID * 4);   // GEMM out (per layer)
  float*  agg   = (float*) carve((size_t)N * DHID * 4);   // scatter accumulator
  __bf16* H3b   = (__bf16*)Hw;                            // overlay Hw+agg (N*256 bf16)
  __bf16* xb    = (__bf16*)carve((size_t)N * FIN * 2);
  __bf16* hb    = (__bf16*)carve((size_t)N * DHID * 2);
  float*  nrm   = (float*) carve((size_t)E * 4);
  float*  dinv  = (float*) carve((size_t)N * 4);
  float*  selfn = (float*) carve((size_t)N * 4);
  float*  deg   = (float*) carve((size_t)N * 4);
  __bf16* W0p   = (__bf16*)carve((size_t)FIN * DHID * 2);
  __bf16* W1p   = (__bf16*)carve((size_t)DHID * DHID * 2);
  __bf16* W2p   = (__bf16*)carve((size_t)DHID * DHID * 2);
  __bf16* Wd1p  = (__bf16*)carve((size_t)DHID * HMLP * 2);
  __bf16* Wd2p  = (__bf16*)carve((size_t)HMLP * 16 * 2);

  auto nb = [](long n) { return (unsigned)((n + 255) / 256); };

  // ---- convert A-side input; pack all weights into B-fragment layout ----
  k_f2bf <<<nb((long)N * FIN), 256, 0, stream>>>(x, xb, (long)N * FIN);
  k_packB<<<nb(FIN * DHID),  256, 0, stream>>>(W0,  W0p,  FIN,  DHID, DHID);
  k_packB<<<nb(DHID * DHID), 256, 0, stream>>>(W1,  W1p,  DHID, DHID, DHID);
  k_packB<<<nb(DHID * DHID), 256, 0, stream>>>(W2,  W2p,  DHID, DHID, DHID);
  k_packB<<<nb(DHID * HMLP), 256, 0, stream>>>(Wd1, Wd1p, DHID, HMLP, HMLP);
  k_packB<<<nb(HMLP * 16),   256, 0, stream>>>(Wd2, Wd2p, HMLP, 16,   6);

  // ---- GCN renormalization ----
  k_deg_init <<<nb(N), 256, 0, stream>>>(deg, N);
  k_deg_accum<<<nb(E), 256, 0, stream>>>(dstI, ew, deg, E);
  k_dinv     <<<nb(N), 256, 0, stream>>>(deg, dinv, selfn, N);
  k_norm     <<<nb(E), 256, 0, stream>>>(srcI, dstI, ew, dinv, nrm, E);

  // NT = n-tiles per wave (A-fragment reuse). Npad/16 must be divisible by NT.
  auto gemm = [&](const __bf16* A, const __bf16* Bf, const float* bias,
                  float* C, __bf16* Cbf, int M, int K, int Npad, int Nreal,
                  int ldc, int relu, int NTsel) {
    if (NTsel == 4) {
      dim3 grid((unsigned)((M / 16 + 3) / 4), (unsigned)(Npad / 64));
      k_gemm<4><<<grid, 128, 0, stream>>>(A, Bf, bias, C, Cbf, M, K, Npad,
                                          Nreal, ldc, relu);
    } else {
      dim3 grid((unsigned)((M / 16 + 3) / 4), (unsigned)(Npad / 16));
      k_gemm<1><<<grid, 128, 0, stream>>>(A, Bf, bias, C, Cbf, M, K, Npad,
                                          Nreal, ldc, relu);
    }
  };

  // ---- one GCN layer: gemm -> self-loop -> edge scatter -> bias+relu ----
  auto gcn = [&](const __bf16* hin, int K, const __bf16* Wp, const float* bias) {
    gemm(hin, Wp, nullptr, Hw, nullptr, N, K, DHID, DHID, DHID, 0, 4);
    const long nd = (long)N * DHID;
    k_selfinit<<<nb(nd), 256, 0, stream>>>(Hw, selfn, agg, nd);
    k_scatter <<<nb((long)E * 16), 256, 0, stream>>>(Hw, nrm, srcI, dstI, agg, E);
    k_biasrelu<<<nb(nd), 256, 0, stream>>>(agg, bias, hb, nd);
  };

  gcn(xb, FIN,  W0p, b0);
  gcn(hb, DHID, W1p, b1);
  gcn(hb, DHID, W2p, b2);

  // ---- MLP head ----
  // H3 = relu(h @ Wd1 + bd1), written bf16 directly (overlays dead Hw+agg)
  gemm(hb, Wd1p, bd1, nullptr, H3b, N, DHID, HMLP, HMLP, HMLP, 1, 4);
  // out = H3 @ Wd2 + bd2 ; B packed/padded to 16 cols, store guarded to 6
  gemm(H3b, Wd2p, bd2, (float*)d_out, nullptr, N, HMLP, 16, 6, 6, 0, 1);
}
